// BiAttention_21766894256451
// MI455X (gfx1250) — matrix operand order repaired and checked by reading, not compiled, not run on hardware
//
#include <hip/hip_runtime.h>
#include <hip/hip_bf16.h>

#define B_ 8
#define L_ 2048
#define H_ 1024
#define JL 64
#define T_ (L_ - JL)      // 1984
#define NTILE (T_ / 16)   // 124

typedef __attribute__((ext_vector_type(16))) __bf16 v16bf;
typedef __attribute__((ext_vector_type(8)))  float  v8f;

// LDS partition (bytes)
#define SM_QJH   (JL * H_ * 2)       // query [j][h] bf16 : 131072
#define SM_QHJ   (H_ * JL * 2)       // query [h][j] bf16 : 131072
#define SM_QTERM (JL * 4)            // q_term[j]         : 256
#define SM_PLDS  (4 * 16 * JL * 4)   // per-wave 16x64 f32 scratch : 16384
#define SMEM_BYTES (SM_QJH + SM_QHJ + SM_QTERM + SM_PLDS)

// ---------------------------------------------------------------------------
// Fused: S = c_term + q_term + (context*w_m)@query^T ; softmax over j ;
// c2q = P@query ; writes query-out, rowmax, and G segments 0..2 (coalesced).
// Block = 128 threads (4 waves); each wave owns one 16-row context tile.
// ---------------------------------------------------------------------------
__global__ void __launch_bounds__(128)
bi_attn_main(const float* __restrict__ enc, const float* __restrict__ w,
             float* __restrict__ out_query, float* __restrict__ out_G,
             float* __restrict__ rowmax_ws)
{
    extern __shared__ __align__(64) unsigned char smem[];
    __bf16* q_jh   = (__bf16*)smem;
    __bf16* q_hj   = (__bf16*)(smem + SM_QJH);
    float*  q_term = (float*)(smem + SM_QJH + SM_QHJ);
    float*  p_lds  = (float*)(smem + SM_QJH + SM_QHJ + SM_QTERM);

    const int b    = blockIdx.y;
    const int tid  = threadIdx.x;
    const int wave = tid >> 5;
    const int lane = tid & 31;
    const int half = lane >> 4;   // 0: lanes 0-15, 1: lanes 16-31
    const int mrow = lane & 15;

    const float* wc = w;
    const float* wq = w + H_;
    const float* wm = w + 2 * H_;
    const float* encb = enc + (size_t)b * L_ * H_;

    // ---- stage query in bf16 (both layouts); block x==0 also emits query out
    for (int idx = tid; idx < JL * H_; idx += 128) {
        int j = idx >> 10;
        int h = idx & (H_ - 1);
        float v = encb[idx];
        q_jh[j * H_ + h] = (__bf16)v;
        q_hj[h * JL + j] = (__bf16)v;
        if (blockIdx.x == 0) out_query[(size_t)b * JL * H_ + idx] = v;
    }
    __syncthreads();
    // q_term[j] = dot(query[j], w_q): 128 threads -> 2 half-row partials in p_lds
    {
        int j = tid & (JL - 1);
        int seg = tid >> 6;                 // 0 or 1
        float s = 0.f;
        for (int h = seg * (H_ / 2); h < (seg + 1) * (H_ / 2); ++h)
            s += (float)q_jh[j * H_ + h] * wq[h];
        p_lds[seg * JL + j] = s;
    }
    __syncthreads();
    if (tid < JL) q_term[tid] = p_lds[tid] + p_lds[JL + tid];
    __syncthreads();

    // ---- this wave's tile of context rows
    const int tt = blockIdx.x * 4 + wave;   // 0..123
    const int t0 = tt * 16;
    const float* crow = encb + (size_t)(JL + t0 + mrow) * H_;

    v8f accS[4] = {};                       // S tile 16 x 64 (4 n-subtiles)
    float csum = 0.f;                       // partial c_term for row mrow

    for (int kb = 0; kb < H_ / 32; ++kb) {  // K over H, 32 per WMMA
        const int kbase = kb * 32;
        if (kbase + 256 < H_)
            __builtin_prefetch(crow + kbase + 256, 0, 1);  // global_prefetch
        v16bf afrag;
        #pragma unroll
        for (int v = 0; v < 8; ++v) {
            int h = kbase + (half << 4) + (v << 1);
            float x0 = crow[h], x1 = crow[h + 1];
            afrag[2 * v]     = (__bf16)(x0 * wm[h]);
            afrag[2 * v + 1] = (__bf16)(x1 * wm[h + 1]);
            csum += x0 * wc[h] + x1 * wc[h + 1];
        }
        #pragma unroll
        for (int nj = 0; nj < 4; ++nj) {
            int j = nj * 16 + mrow;         // B-operand column
            v16bf bfrag = *(const v16bf*)(q_jh + j * H_ + kbase + half * 16);
            accS[nj] = __builtin_amdgcn_wmma_f32_16x16x32_bf16(
                false, afrag, false, bfrag, (short)0, accS[nj], false, false);
        }
    }
    csum += __shfl_xor(csum, 16, 32);       // full c_term for row mrow

    float qt[4];
    #pragma unroll
    for (int nj = 0; nj < 4; ++nj) qt[nj] = q_term[nj * 16 + mrow];

    // ---- S = acc + c_term + q_term ; softmax over j ; record rowmax
    #pragma unroll
    for (int r = 0; r < 8; ++r) {
        float ct = __shfl(csum, half * 8 + r, 32);  // row = half*8 + r
        float rm = -3.4e38f;
        #pragma unroll
        for (int nj = 0; nj < 4; ++nj) {
            float s = accS[nj][r] + ct + qt[nj];
            accS[nj][r] = s;
            rm = fmaxf(rm, s);
        }
        #pragma unroll
        for (int off = 8; off >= 1; off >>= 1)
            rm = fmaxf(rm, __shfl_xor(rm, off, 32));
        float rs = 0.f;
        #pragma unroll
        for (int nj = 0; nj < 4; ++nj) {
            float e = __expf(accS[nj][r] - rm);
            accS[nj][r] = e;
            rs += e;
        }
        #pragma unroll
        for (int off = 8; off >= 1; off >>= 1)
            rs += __shfl_xor(rs, off, 32);
        float inv = 1.0f / rs;
        #pragma unroll
        for (int nj = 0; nj < 4; ++nj) accS[nj][r] *= inv;
        if (mrow == 0)
            rowmax_ws[(size_t)b * T_ + t0 + half * 8 + r] = rm;
    }

    // ---- round-trip P through LDS to build WMMA A-fragments
    float* pw = p_lds + wave * (16 * JL);
    #pragma unroll
    for (int r = 0; r < 8; ++r) {
        #pragma unroll
        for (int nj = 0; nj < 4; ++nj)
            pw[(half * 8 + r) * JL + nj * 16 + mrow] = accS[nj][r];
    }
    __syncthreads();

    v16bf pa[2];
    #pragma unroll
    for (int ks = 0; ks < 2; ++ks) {
        #pragma unroll
        for (int v = 0; v < 8; ++v) {
            int jb = ks * 32 + (half << 4) + (v << 1);
            pa[ks][2 * v]     = (__bf16)pw[mrow * JL + jb];
            pa[ks][2 * v + 1] = (__bf16)pw[mrow * JL + jb + 1];
        }
    }

    // ---- c2q = P @ query, 64-wide h chunks; emit G segs 0..2 coalesced
    const size_t gbase = ((size_t)b * T_ + t0) * (size_t)(4 * H_);
    for (int hc = 0; hc < H_; hc += 64) {
        // 4 WMMA n-tiles -> stage 16x64 c2q chunk into this wave's pw
        #pragma unroll
        for (int nt4 = 0; nt4 < 4; ++nt4) {
            v8f accC = {};
            int hcol = hc + nt4 * 16 + mrow;
            #pragma unroll
            for (int ks = 0; ks < 2; ++ks) {
                v16bf bfrag = *(const v16bf*)(q_hj + hcol * JL + ks * 32 + half * 16);
                accC = __builtin_amdgcn_wmma_f32_16x16x32_bf16(
                    false, pa[ks], false, bfrag, (short)0, accC, false, false);
            }
            #pragma unroll
            for (int r = 0; r < 8; ++r)
                pw[(half * 8 + r) * 64 + nt4 * 16 + mrow] = accC[r];
        }
        // coalesced emit: 16 rows x 64 cols, float4 per lane (b128 traffic)
        #pragma unroll
        for (int i = 0; i < 8; ++i) {
            int f   = i * 128 + lane * 4;   // flat index into 16x64 chunk
            int row = f >> 6;               // 0..15
            int col = f & 63;
            float4 c2q4 = *(const float4*)(pw + row * 64 + col);
            float4 ctx4 = *(const float4*)(encb + (size_t)(JL + t0 + row) * H_ + hc + col);
            float* grow = out_G + gbase + (size_t)row * (4 * H_) + hc + col;
            *(float4*)(grow)           = ctx4;
            *(float4*)(grow + H_)      = c2q4;
            *(float4*)(grow + 2 * H_)  = make_float4(ctx4.x * c2q4.x, ctx4.y * c2q4.y,
                                                     ctx4.z * c2q4.z, ctx4.w * c2q4.w);
        }
    }
}

// ---------------------------------------------------------------------------
// Per-batch: b_att = softmax_T(rowmax), in place.
// ---------------------------------------------------------------------------
__global__ void __launch_bounds__(256)
batt_reduce(float* __restrict__ rowmax_ws)
{
    __shared__ float red[256];
    const int b = blockIdx.x, tid = threadIdx.x;
    float* rm = rowmax_ws + (size_t)b * T_;
    float m = -3.4e38f;
    for (int t = tid; t < T_; t += 256) m = fmaxf(m, rm[t]);
    red[tid] = m; __syncthreads();
    for (int s = 128; s > 0; s >>= 1) {
        if (tid < s) red[tid] = fmaxf(red[tid], red[tid + s]);
        __syncthreads();
    }
    float M = red[0]; __syncthreads();
    float z = 0.f;
    for (int t = tid; t < T_; t += 256) z += __expf(rm[t] - M);
    red[tid] = z; __syncthreads();
    for (int s = 128; s > 0; s >>= 1) {
        if (tid < s) red[tid] += red[tid + s];
        __syncthreads();
    }
    float invZ = 1.0f / red[0]; __syncthreads();
    for (int t = tid; t < T_; t += 256) rm[t] = __expf(rm[t] - M) * invZ;
}

// ---------------------------------------------------------------------------
// q2c[b,h] = sum_t b_att[b,t] * context[b,t,h]
// ---------------------------------------------------------------------------
__global__ void __launch_bounds__(256)
q2c_kernel(const float* __restrict__ enc, const float* __restrict__ batt,
           float* __restrict__ q2c)
{
    const int b = blockIdx.y;
    const int h = blockIdx.x * 256 + threadIdx.x;
    const float* encb = enc + ((size_t)b * L_ + JL) * H_;
    const float* ba = batt + (size_t)b * T_;
    float acc = 0.f;
    for (int t = 0; t < T_; ++t) acc += ba[t] * encb[(size_t)t * H_ + h];
    q2c[b * H_ + h] = acc;
}

// ---------------------------------------------------------------------------
// G segment 3: context * q2c (broadcast over t), float4 coalesced
// ---------------------------------------------------------------------------
__global__ void __launch_bounds__(256)
g3_kernel(const float* __restrict__ enc, const float* __restrict__ q2c,
          float* __restrict__ out_G)
{
    int idx = (blockIdx.x * 256 + threadIdx.x) * 4;   // < B*T*H, 4 per thread
    int h   = idx & (H_ - 1);
    int tmp = idx >> 10;
    int t   = tmp % T_;
    int b   = tmp / T_;
    float4 ctx = *(const float4*)(enc + ((size_t)b * L_ + JL + t) * H_ + h);
    float4 qc  = *(const float4*)(q2c + b * H_ + h);
    *(float4*)(out_G + ((size_t)b * T_ + t) * (size_t)(4 * H_) + 3 * H_ + h) =
        make_float4(ctx.x * qc.x, ctx.y * qc.y, ctx.z * qc.z, ctx.w * qc.w);
}

extern "C" void kernel_launch(void* const* d_in, const int* in_sizes, int n_in,
                              void* d_out, int out_size, void* d_ws, size_t ws_size,
                              hipStream_t stream)
{
    const float* enc = (const float*)d_in[0];
    const float* w   = (const float*)d_in[1];
    (void)in_sizes; (void)n_in; (void)out_size; (void)ws_size;

    float* out       = (float*)d_out;
    float* out_query = out;                                // B*JL*H floats
    float* out_G     = out + (size_t)B_ * JL * H_;         // B*T*4H floats

    float* rowmax_ws = (float*)d_ws;                       // B*T floats -> b_att
    float* q2c_ws    = rowmax_ws + (size_t)B_ * T_;        // B*H floats

    bi_attn_main<<<dim3(NTILE / 4, B_), 128, SMEM_BYTES, stream>>>(
        enc, w, out_query, out_G, rowmax_ws);
    batt_reduce<<<dim3(B_), 256, 0, stream>>>(rowmax_ws);
    q2c_kernel<<<dim3(H_ / 256, B_), 256, 0, stream>>>(enc, rowmax_ws, q2c_ws);
    g3_kernel<<<dim3((B_ * T_ * H_) / (256 * 4)), 256, 0, stream>>>(enc, q2c_ws, out_G);
}
